// Linear8bit_54425825575255
// MI455X (gfx1250) — 8-bit matrix operand order repaired and checked by reading, compiled, not run on hardware
//
#include <hip/hip_runtime.h>
#include <stdint.h>


typedef __attribute__((ext_vector_type(8))) int v8i;
typedef __attribute__((ext_vector_type(4))) int v4i;
typedef __attribute__((ext_vector_type(2))) int v2i;

#define BM 128
#define BN 256
#define BK 64
#define LDA 80
#define LDB 80

#if defined(__has_builtin)
#if __has_builtin(__builtin_amdgcn_global_load_async_to_lds_b128)
#define HAVE_ASYNC_LDS 1
#endif
#if __has_builtin(__builtin_amdgcn_s_wait_asynccnt)
#define HAVE_WAIT_ASYNC 1
#endif
#endif

typedef __attribute__((address_space(1))) v4i* gv4_t;
typedef __attribute__((address_space(3))) v4i* lv4_t;

__device__ __forceinline__ void wait_asynccnt0() {
#if defined(HAVE_WAIT_ASYNC)
    __builtin_amdgcn_s_wait_asynccnt(0);
#else
    asm volatile("s_wait_asynccnt 0x0" ::: "memory");
#endif
}

__global__ __launch_bounds__(256)
void i8gemm_dequant_kernel(const int8_t* __restrict__ xq,
                           const float*  __restrict__ sx,
                           const int8_t* __restrict__ wq,
                           const float*  __restrict__ wsc,
                           const float*  __restrict__ bias,
                           float*        __restrict__ out,
                           int R, int N, int K)
{
    __shared__ __align__(16) int8_t As[2][BM * LDA];
    __shared__ __align__(16) int8_t Bs[2][BN * LDB];

    const int tid      = threadIdx.x;
    const int lane     = tid & 31;
    const int wave     = tid >> 5;
    const int laneLow  = lane & 15;
    const int laneHalf = lane >> 4;

    const int waveM = wave & 1;
    const int waveN = wave >> 1;
    const int wm    = waveM * 64;
    const int wn    = waveN * 64;

    const int rowBase = blockIdx.y * BM;
    const int colBase = blockIdx.x * BN;

    v8i acc[4][4];
#pragma unroll
    for (int mt = 0; mt < 4; ++mt)
#pragma unroll
        for (int nt = 0; nt < 4; ++nt)
#pragma unroll
            for (int j = 0; j < 8; ++j)
                acc[mt][nt][j] = 0;

    auto stage = [&](int buf, int kt) {
#if defined(HAVE_ASYNC_LDS)
#pragma unroll
        for (int it = 0; it < 2; ++it) {
            const int c = tid + it * 256, row = c >> 2, ko = (c & 3) * 16;
            __builtin_amdgcn_global_load_async_to_lds_b128(
                (gv4_t)(xq + (size_t)(rowBase + row) * (size_t)K + (size_t)(kt + ko)),
                (lv4_t)(&As[buf][row * LDA + ko]), 0, 0);
        }
#pragma unroll
        for (int it = 0; it < 4; ++it) {
            const int c = tid + it * 256, row = c >> 2, ko = (c & 3) * 16;
            __builtin_amdgcn_global_load_async_to_lds_b128(
                (gv4_t)(wq + (size_t)(colBase + row) * (size_t)K + (size_t)(kt + ko)),
                (lv4_t)(&Bs[buf][row * LDB + ko]), 0, 0);
        }
#else
        v4i va[2], vb[4];
#pragma unroll
        for (int it = 0; it < 2; ++it) {
            const int c = tid + it * 256, row = c >> 2, ko = (c & 3) * 16;
            va[it] = *(const v4i*)(xq + (size_t)(rowBase + row) * (size_t)K + (size_t)(kt + ko));
        }
#pragma unroll
        for (int it = 0; it < 4; ++it) {
            const int c = tid + it * 256, row = c >> 2, ko = (c & 3) * 16;
            vb[it] = *(const v4i*)(wq + (size_t)(colBase + row) * (size_t)K + (size_t)(kt + ko));
        }
#pragma unroll
        for (int it = 0; it < 2; ++it) {
            const int c = tid + it * 256, row = c >> 2, ko = (c & 3) * 16;
            *(v4i*)(&As[buf][row * LDA + ko]) = va[it];
        }
#pragma unroll
        for (int it = 0; it < 4; ++it) {
            const int c = tid + it * 256, row = c >> 2, ko = (c & 3) * 16;
            *(v4i*)(&Bs[buf][row * LDB + ko]) = vb[it];
        }
#endif
    };

    stage(0, 0);
    wait_asynccnt0();
    __syncthreads();

    const int nk = K / BK;
    for (int ki = 0; ki < nk; ++ki) {
        const int cur = ki & 1;

        if (ki + 1 < nk)
            stage(cur ^ 1, (ki + 1) * BK);

        v8i afrag[4];
#pragma unroll
        for (int mt = 0; mt < 4; ++mt) {
            const int8_t* ap = &As[cur][(wm + mt * 16 + laneLow) * LDA + laneHalf * 16];
#pragma unroll
            for (int i = 0; i < 4; ++i) {
                v2i t = *(const v2i*)(ap + (i >> 1) * 32 + (i & 1) * 8);
                afrag[mt][2 * i]     = t[0];
                afrag[mt][2 * i + 1] = t[1];
            }
        }

        v8i bfrag[4];
#pragma unroll
        for (int nt = 0; nt < 4; ++nt) {
            const int8_t* bp = &Bs[cur][(wn + nt * 16 + laneLow) * LDB + laneHalf * 16];
            v4i t0 = *(const v4i*)(bp);
            v4i t1 = *(const v4i*)(bp + 32);
#pragma unroll
            for (int j = 0; j < 4; ++j) {
                bfrag[nt][j]     = t0[j];
                bfrag[nt][4 + j] = t1[j];
            }
        }

#pragma unroll
        for (int mt = 0; mt < 4; ++mt)
#pragma unroll
            for (int nt = 0; nt < 4; ++nt)
                acc[mt][nt] = __builtin_amdgcn_wmma_i32_16x16x64_iu8(
                     true, afrag[mt],
                     true, bfrag[nt],
                    acc[mt][nt],
                     false,  false);

        if (ki + 1 < nk) {
            wait_asynccnt0();
            __syncthreads();
        }
    }

#pragma unroll
    for (int mt = 0; mt < 4; ++mt) {
        const int r0 = rowBase + wm + mt * 16 + laneHalf * 8;
        float sxv[8];
#pragma unroll
        for (int v = 0; v < 8; ++v)
            sxv[v] = sx[r0 + v];

#pragma unroll
        for (int nt = 0; nt < 4; ++nt) {
            const int col = colBase + wn + nt * 16 + laneLow;
            const float wsv = wsc[col];
            const float bv  = bias[col];
#pragma unroll
            for (int v = 0; v < 8; ++v) {
                out[(size_t)(r0 + v) * (size_t)N + (size_t)col] =
                    (float)acc[mt][nt][v] * sxv[v] * wsv + bv;
            }
        }
    }
}

extern "C" void kernel_launch(void* const* d_in, const int* in_sizes, int n_in,
                              void* d_out, int out_size, void* d_ws, size_t ws_size,
                              hipStream_t stream) {
    const int8_t* xq   = (const int8_t*)d_in[0];
    const float*  sx   = (const float*)d_in[1];
    const int8_t* wq   = (const int8_t*)d_in[2];
    const float*  wsc  = (const float*)d_in[3];
    const float*  bias = (const float*)d_in[4];
    float* out = (float*)d_out;

    const int R = in_sizes[1];
    const int N = in_sizes[3];
    const int K = in_sizes[2] / N;

    dim3 grid(N / BN, R / BM);
    i8gemm_dequant_kernel<<<grid, 256, 0, stream>>>(xq, sx, wq, wsc, bias, out, R, N, K);
}
